// LSTM_61220463837533
// MI455X (gfx1250) — matrix operand order repaired and checked by reading, not compiled, not run on hardware
//
#include <hip/hip_runtime.h>

// ---------------------------------------------------------------- constants
#define HID  1024
#define VOC  512
#define BAT  128
#define SEQ  512
#define KTOT (HID + VOC)   // 1536
#define GM   (4 * HID)     // 4096 gate rows (interleaved: row = 4*h + gate)
#define NROW ((size_t)BAT * SEQ)  // 65536 logits rows / Xg columns (c = t*B + b for Xg)

typedef __bf16 bf16;
typedef __attribute__((ext_vector_type(16))) __bf16 v16bf;
typedef __attribute__((ext_vector_type(8)))  float  v8f;

union FragBF { uint4 q[2]; v16bf v; };
union Pack8  { uint4 q;    bf16 h[8]; };

// ------------------------------------------------ one K=32 step of a 64x32 tile
// A: row-major [M x lda] bf16 (weights).  Bm: column-major [N cols x ldb] bf16.
__device__ __forceinline__ void mma_k32(v8f (&acc)[4][2],
    const bf16* __restrict__ A,  size_t lda, int mBase, int k0A,
    const bf16* __restrict__ Bm, size_t ldb, int nBase, int k0B,
    int l16, int half)
{
    FragBF bfr[2];
#pragma unroll
    for (int j = 0; j < 2; ++j) {
        const uint4* p = (const uint4*)(Bm + (size_t)(nBase + j*16 + l16)*ldb
                                           + k0B + half*16);
        bfr[j].q[0] = p[0];
        bfr[j].q[1] = p[1];
    }
    FragBF afr[4];
#pragma unroll
    for (int i = 0; i < 4; ++i) {
        const bf16* row = A + (size_t)(mBase + i*16 + l16)*lda + k0A + half*16;
        afr[i].q[0] = *(const uint4*)(row);
        afr[i].q[1] = *(const uint4*)(row + 8);
    }
#pragma unroll
    for (int i = 0; i < 4; ++i)
#pragma unroll
        for (int j = 0; j < 2; ++j)
            acc[i][j] = __builtin_amdgcn_wmma_f32_16x16x32_bf16(
                false, afr[i].v, false, bfr[j].v, (short)0, acc[i][j],
                false, false);
}

__device__ __forceinline__ float sigmoidf_(float x) {
    return 1.0f / (1.0f + __expf(-x));
}

// gate epilogue shared by both step kernels (rows r = 4h+gate -> lane-local)
__device__ __forceinline__ void gate_epilogue(v8f (&acc)[4][2],
    const float* __restrict__ bg, float* __restrict__ c,
    bf16* __restrict__ Aall, int tileM, int tileN, int l16, int half, int t)
{
#pragma unroll
    for (int i = 0; i < 4; ++i) {
        const int m0 = tileM + i * 16;
#pragma unroll
        for (int j = 0; j < 2; ++j) {
            const int n = tileN + j * 16 + l16;
#pragma unroll
            for (int p = 0; p < 2; ++p) {
                const int h = (m0 >> 2) + half * 2 + p;
                const int r = 4 * h;
                float gf = acc[i][j][4*p + 0] + bg[r + 0];
                float gu = acc[i][j][4*p + 1] + bg[r + 1];
                float gc = acc[i][j][4*p + 2] + bg[r + 2];
                float go = acc[i][j][4*p + 3] + bg[r + 3];
                float f  = sigmoidf_(gf);
                float u  = sigmoidf_(gu);
                float cc = tanhf(gc);
                float o  = sigmoidf_(go);
                size_t idx = (size_t)n * HID + h;
                float cn = f * c[idx] + u * cc;
                c[idx] = cn;
                float an = o * tanhf(cn);
                Aall[((size_t)n * SEQ + t) * HID + h] = (bf16)an;
            }
        }
    }
}

// ---------------------------------------------------------------- setup kernels
__global__ void conv_x_kernel(const float* __restrict__ X, bf16* __restrict__ xb)
{
    // xb[t*B*V + b*V + v] = X[b*T*V + t*V + v]   (column-major: col = (t,b))
    size_t i = (size_t)blockIdx.x * blockDim.x + threadIdx.x;
    if (i >= (size_t)SEQ * BAT * VOC) return;
    int v = (int)(i % VOC);
    int b = (int)((i / VOC) % BAT);
    int t = (int)(i / ((size_t)VOC * BAT));
    xb[i] = (bf16)X[(size_t)b * SEQ * VOC + (size_t)t * VOC + v];
}

__global__ void pack_wg_kernel(const float* __restrict__ Wf,
                               const float* __restrict__ Wu,
                               const float* __restrict__ Wc,
                               const float* __restrict__ Wo,
                               bf16* __restrict__ Wg)
{
    size_t i = (size_t)blockIdx.x * blockDim.x + threadIdx.x;
    if (i >= (size_t)GM * KTOT) return;
    int k = (int)(i % KTOT);
    int r = (int)(i / KTOT);
    int h = r >> 2, g = r & 3;
    const float* src = (g == 0) ? Wf : (g == 1) ? Wu : (g == 2) ? Wc : Wo;
    Wg[i] = (bf16)src[(size_t)h * KTOT + k];
}

__global__ void pack_bias_kernel(const float* __restrict__ bf_,
                                 const float* __restrict__ bu_,
                                 const float* __restrict__ bc_,
                                 const float* __restrict__ bo_,
                                 float* __restrict__ bg)
{
    int r = blockIdx.x * blockDim.x + threadIdx.x;
    if (r >= GM) return;
    int h = r >> 2, g = r & 3;
    const float* src = (g == 0) ? bf_ : (g == 1) ? bu_ : (g == 2) ? bc_ : bo_;
    bg[r] = src[h];
}

__global__ void pack_wy_kernel(const float* __restrict__ Wy, bf16* __restrict__ Wyb)
{
    size_t i = (size_t)blockIdx.x * blockDim.x + threadIdx.x;
    if (i < (size_t)VOC * HID) Wyb[i] = (bf16)Wy[i];
}

__global__ void pack_ac_kernel(const float* __restrict__ a_prev,
                               const float* __restrict__ c_prev,
                               bf16* __restrict__ a0, float* __restrict__ c0)
{
    size_t i = (size_t)blockIdx.x * blockDim.x + threadIdx.x;
    if (i >= (size_t)BAT * HID) return;
    int b = (int)(i / HID);
    int h = (int)(i % HID);
    a0[i] = (bf16)a_prev[(size_t)h * BAT + b];   // column-major (col = batch)
    c0[i] = c_prev[(size_t)h * BAT + b];
}

// ---------------------------------------------------------------- Xg precompute
// Xg(4096 x 65536) = Wx(4096x512) @ Xall(512 x 65536), col c = t*B + b.
// Fully parallel: hoists the input-projection off the sequential critical path.
__global__ void __launch_bounds__(256)
xg_kernel(const bf16* __restrict__ Wg,    // [4096 x 1536] rm; Wx = cols 1024..1535
          const bf16* __restrict__ xb,    // [(T*B) cols x 512] cm
          bf16*       __restrict__ Xgb)   // [(T*B) cols x 4096] cm
{
    const int lane = threadIdx.x & 31;
    const int wave = threadIdx.x >> 5;
    const int gw   = blockIdx.x * 8 + wave;     // 0..131071
    const int tileM = (gw & 63) * 64;           // 64 row-tiles
    const int tileN = (gw >> 6) * 32;           // 2048 col-tiles
    const int half = lane >> 4;
    const int l16  = lane & 15;

    v8f acc[4][2];
    v8f zero = {0.f, 0.f, 0.f, 0.f, 0.f, 0.f, 0.f, 0.f};
#pragma unroll
    for (int i = 0; i < 4; ++i)
#pragma unroll
        for (int j = 0; j < 2; ++j) acc[i][j] = zero;

#pragma unroll 1
    for (int k0 = 0; k0 < VOC; k0 += 32)
        mma_k32(acc, Wg, KTOT, tileM, HID + k0, xb, VOC, tileN, k0, l16, half);

#pragma unroll
    for (int i = 0; i < 4; ++i) {
        const int m0 = tileM + i * 16;
#pragma unroll
        for (int j = 0; j < 2; ++j) {
            const int cidx = tileN + j * 16 + l16;
            Pack8 pk;
#pragma unroll
            for (int q = 0; q < 8; ++q) pk.h[q] = (bf16)acc[i][j][q];
            *(uint4*)(Xgb + (size_t)cidx * GM + m0 + half * 8) = pk.q;
        }
    }
}

// ---------------------------------------------------------------- LSTM steps
// Split path: acc initialized from precomputed Xg; only recurrent K=1024 runs.
__global__ void __launch_bounds__(256)
lstm_step_pre_kernel(const bf16*  __restrict__ Wg,
                     const float* __restrict__ bg,
                     const bf16*  __restrict__ Xgb,     // [(T*B) cols x 4096] cm
                     const bf16*  __restrict__ a_base,  // col n at a_base + n*a_ld
                     size_t a_ld,
                     float*       __restrict__ c,
                     bf16*        __restrict__ Aall,
                     int t)
{
    const int lane = threadIdx.x & 31;
    const int wave = threadIdx.x >> 5;
    const int gw   = blockIdx.x * 8 + wave;     // 0..255
    const int tileM = (gw >> 2) * 64;
    const int tileN = (gw & 3)  * 32;
    const int half = lane >> 4;
    const int l16  = lane & 15;

    v8f acc[4][2];
#pragma unroll
    for (int i = 0; i < 4; ++i) {
        const int m0 = tileM + i * 16;
#pragma unroll
        for (int j = 0; j < 2; ++j) {
            const int n = tileN + j * 16 + l16;
            Pack8 pk;
            pk.q = *(const uint4*)(Xgb + ((size_t)t * BAT + n) * GM + m0 + half * 8);
#pragma unroll
            for (int q = 0; q < 8; ++q) acc[i][j][q] = (float)pk.h[q];
        }
    }

#pragma unroll 1
    for (int k0 = 0; k0 < HID; k0 += 32)
        mma_k32(acc, Wg, KTOT, tileM, k0, a_base, a_ld, tileN, k0, l16, half);

    gate_epilogue(acc, bg, c, Aall, tileM, tileN, l16, half, t);
}

// Fused fallback: full K=1536 per step (used if workspace can't hold Xg).
__global__ void __launch_bounds__(256)
lstm_step_fused_kernel(const bf16*  __restrict__ Wg,
                       const float* __restrict__ bg,
                       const bf16*  __restrict__ xt,      // [B cols x V] cm
                       const bf16*  __restrict__ a_base,
                       size_t a_ld,
                       float*       __restrict__ c,
                       bf16*        __restrict__ Aall,
                       int t)
{
    const int lane = threadIdx.x & 31;
    const int wave = threadIdx.x >> 5;
    const int gw   = blockIdx.x * 8 + wave;
    const int tileM = (gw >> 2) * 64;
    const int tileN = (gw & 3)  * 32;
    const int half = lane >> 4;
    const int l16  = lane & 15;

    v8f acc[4][2];
    v8f zero = {0.f, 0.f, 0.f, 0.f, 0.f, 0.f, 0.f, 0.f};
#pragma unroll
    for (int i = 0; i < 4; ++i)
#pragma unroll
        for (int j = 0; j < 2; ++j) acc[i][j] = zero;

#pragma unroll 1
    for (int k0 = 0; k0 < HID; k0 += 32)
        mma_k32(acc, Wg, KTOT, tileM, k0, a_base, a_ld, tileN, k0, l16, half);
#pragma unroll 1
    for (int k0 = 0; k0 < VOC; k0 += 32)
        mma_k32(acc, Wg, KTOT, tileM, HID + k0, xt, VOC, tileN, k0, l16, half);

    gate_epilogue(acc, bg, c, Aall, tileM, tileN, l16, half, t);
}

// ---------------------------------------------------------------- output GEMM
// logits(512 x 65536) = Wy(512x1024) @ Aall(1024 x 65536); out[col*512 + v]
__global__ void __launch_bounds__(256)
logits_kernel(const bf16*  __restrict__ Wyb,   // [512 x 1024] rm
              const float* __restrict__ by,    // [512]
              const bf16*  __restrict__ Aall,  // [(B*T) cols x 1024] cm
              float*       __restrict__ out)   // [(B*T) x 512] rm
{
    const int lane = threadIdx.x & 31;
    const int wave = threadIdx.x >> 5;
    const int gw   = blockIdx.x * 8 + wave;     // 0..16383
    const int tileM = (gw & 7) * 64;
    const int tileN = (gw >> 3) * 32;
    const int half = lane >> 4;
    const int l16  = lane & 15;

    v8f acc[4][2];
    v8f zero = {0.f, 0.f, 0.f, 0.f, 0.f, 0.f, 0.f, 0.f};
#pragma unroll
    for (int i = 0; i < 4; ++i)
#pragma unroll
        for (int j = 0; j < 2; ++j) acc[i][j] = zero;

#pragma unroll 1
    for (int k0 = 0; k0 < HID; k0 += 32)
        mma_k32(acc, Wyb, HID, tileM, k0, Aall, HID, tileN, k0, l16, half);

#pragma unroll
    for (int i = 0; i < 4; ++i) {
        const int m0 = tileM + i * 16;
        const int v0 = m0 + half * 8;
        const float4* pb = (const float4*)(by + v0);
        float4 blo = pb[0], bhi = pb[1];
#pragma unroll
        for (int j = 0; j < 2; ++j) {
            const int col = tileN + j * 16 + l16;
            float* op = out + (size_t)col * VOC + v0;
            float4 r0 = { acc[i][j][0] + blo.x, acc[i][j][1] + blo.y,
                          acc[i][j][2] + blo.z, acc[i][j][3] + blo.w };
            float4 r1 = { acc[i][j][4] + bhi.x, acc[i][j][5] + bhi.y,
                          acc[i][j][6] + bhi.z, acc[i][j][7] + bhi.w };
            ((float4*)op)[0] = r0;
            ((float4*)op)[1] = r1;
        }
    }
}

// ---------------------------------------------------------------- loss
__global__ void __launch_bounds__(256)
loss_rows_kernel(const float* __restrict__ logits,
                 const int*   __restrict__ labels,
                 float*       __restrict__ rowloss)
{
    const int lane = threadIdx.x & 31;
    const int wave = threadIdx.x >> 5;
    const int row  = blockIdx.x * 8 + wave;
    const float* rp = logits + (size_t)row * VOC;

    float mx = -3.402823466e38f;
    for (int i = lane; i < VOC; i += 32) mx = fmaxf(mx, rp[i]);
#pragma unroll
    for (int s = 16; s; s >>= 1) mx = fmaxf(mx, __shfl_xor(mx, s, 32));
    float se = 0.f;
    for (int i = lane; i < VOC; i += 32) se += __expf(rp[i] - mx);
#pragma unroll
    for (int s = 16; s; s >>= 1) se += __shfl_xor(se, s, 32);
    if (lane == 0) {
        int lab = labels[row];
        rowloss[row] = -(rp[lab] - mx - __logf(se));
    }
}

__global__ void __launch_bounds__(256)
loss_reduce_kernel(const float* __restrict__ rowloss, float* __restrict__ loss)
{
    __shared__ float s[256];
    float a = 0.f;
    for (size_t i = threadIdx.x; i < NROW; i += 256) a += rowloss[i];
    s[threadIdx.x] = a;
    __syncthreads();
#pragma unroll
    for (int st = 128; st; st >>= 1) {
        if ((int)threadIdx.x < st) s[threadIdx.x] += s[threadIdx.x + st];
        __syncthreads();
    }
    if (threadIdx.x == 0) *loss = s[0] * (1.0f / (float)NROW);
}

// ---------------------------------------------------------------- launcher
extern "C" void kernel_launch(void* const* d_in, const int* in_sizes, int n_in,
                              void* d_out, int out_size, void* d_ws, size_t ws_size,
                              hipStream_t stream)
{
    (void)in_sizes; (void)n_in; (void)out_size;
    const float* X    = (const float*)d_in[0];
    const int*   Y    = (const int*)  d_in[1];
    const float* Wf   = (const float*)d_in[2];
    const float* bf_  = (const float*)d_in[3];
    const float* Wu   = (const float*)d_in[4];
    const float* bu_  = (const float*)d_in[5];
    const float* Wc   = (const float*)d_in[6];
    const float* bc_  = (const float*)d_in[7];
    const float* Wo   = (const float*)d_in[8];
    const float* bo_  = (const float*)d_in[9];
    const float* Wy   = (const float*)d_in[10];
    const float* by   = (const float*)d_in[11];
    const float* a_in = (const float*)d_in[12];
    const float* c_in = (const float*)d_in[13];

    float* out  = (float*)d_out;
    float* loss = out + NROW * (size_t)VOC;   // last element

    // workspace carve-up (all 256B aligned)
    uint8_t* w   = (uint8_t*)d_ws;
    uint8_t* end = w + ws_size;
    auto take = [&](size_t bytes) {
        uint8_t* p = w;
        w += (bytes + 255) & ~(size_t)255;
        return p;
    };
    bf16*  xb    = (bf16*) take((size_t)SEQ * BAT * VOC * 2);   //  67 MB
    bf16*  Wg    = (bf16*) take((size_t)GM * KTOT * 2);         // 12.6 MB
    float* bg    = (float*)take((size_t)GM * 4);
    bf16*  Wyb   = (bf16*) take((size_t)VOC * HID * 2);
    bf16*  abuf0 = (bf16*) take((size_t)BAT * HID * 2);
    float* cbuf  = (float*)take((size_t)BAT * HID * 4);
    bf16*  Aall  = (bf16*) take(NROW * (size_t)HID * 2);        // 134 MB
    float* rls   = (float*)take(NROW * 4);

    const size_t XG_BYTES = NROW * (size_t)GM * 2;              // 537 MB
    const bool splitPath = ((size_t)(end - w) >= XG_BYTES + 256);
    bf16* Xgb = splitPath ? (bf16*)take(XG_BYTES) : nullptr;

    // --- one-time conversions (bf16 packing, gate-row interleave) ---
    conv_x_kernel <<<(unsigned)(((size_t)SEQ*BAT*VOC + 255) / 256), 256, 0, stream>>>(X, xb);
    pack_wg_kernel<<<(unsigned)(((size_t)GM*KTOT   + 255) / 256), 256, 0, stream>>>(Wf, Wu, Wc, Wo, Wg);
    pack_bias_kernel<<<(GM + 255) / 256, 256, 0, stream>>>(bf_, bu_, bc_, bo_, bg);
    pack_wy_kernel<<<(unsigned)(((size_t)VOC*HID + 255) / 256), 256, 0, stream>>>(Wy, Wyb);
    pack_ac_kernel<<<((BAT * HID) + 255) / 256, 256, 0, stream>>>(a_in, c_in, abuf0, cbuf);

    if (splitPath) {
        // hoist input projection for all timesteps off the critical path
        xg_kernel<<<16384, 256, 0, stream>>>(Wg, xb, Xgb);
        for (int t = 0; t < SEQ; ++t) {
            const bf16* a_base = (t == 0) ? abuf0 : (Aall + (size_t)(t - 1) * HID);
            size_t      a_ld   = (t == 0) ? (size_t)HID : (size_t)SEQ * HID;
            lstm_step_pre_kernel<<<32, 256, 0, stream>>>(
                Wg, bg, Xgb, a_base, a_ld, cbuf, Aall, t);
        }
    } else {
        for (int t = 0; t < SEQ; ++t) {
            const bf16* xt     = xb + (size_t)t * BAT * VOC;
            const bf16* a_base = (t == 0) ? abuf0 : (Aall + (size_t)(t - 1) * HID);
            size_t      a_ld   = (t == 0) ? (size_t)HID : (size_t)SEQ * HID;
            lstm_step_fused_kernel<<<32, 256, 0, stream>>>(
                Wg, bg, xt, a_base, a_ld, cbuf, Aall, t);
        }
    }

    // --- fused output projection over all T*B columns ---
    logits_kernel<<<2048, 256, 0, stream>>>(Wyb, by, Aall, out);

    // --- softmax-NLL, deterministic two-stage reduce ---
    loss_rows_kernel<<<(unsigned)(NROW / 8), 256, 0, stream>>>(out, Y, rls);
    loss_reduce_kernel<<<1, 256, 0, stream>>>(rls, loss);
}
